// SimCSELoss_47038481826438
// MI455X (gfx1250) — matrix operand order repaired and checked by reading, not compiled, not run on hardware
//
#include <hip/hip_runtime.h>
#include <hip/hip_bf16.h>
#include <stdint.h>

typedef __bf16 bf16;
typedef __bf16 bf16x8  __attribute__((ext_vector_type(8)));
typedef __bf16 bf16x16 __attribute__((ext_vector_type(16)));
typedef float  f32x8   __attribute__((ext_vector_type(8)));

#define N_ROWS 4096
#define DIM    1024
#define ZROWS  8192
#define BM 64
#define BN 128
#define KB 32
#define NKSTEP (DIM / KB)      /* 32 */
#define COLCHUNKS (ZROWS / 32) /* 256 */

// Generic address of __shared__ is {aperture_hi32, lds_offset32}; truncation
// yields the 32-bit LDS byte address the async-load VDST operand wants.
static __device__ __forceinline__ uint32_t lds_addr(const void* p) {
  return (uint32_t)(uintptr_t)p;
}

// ASYNCcnt-tracked DMA: 16B per lane, global (saddr + 32b voffset) -> LDS.
static __device__ __forceinline__ void async_copy_b128(uint32_t lds,
                                                       const void* sbase,
                                                       uint32_t voff) {
  asm volatile("global_load_async_to_lds_b128 %0, %1, %2"
               :: "v"(lds), "v"(voff), "s"(sbase)
               : "memory");
}

static __device__ __forceinline__ bf16x16 comb(bf16x8 a, bf16x8 b) {
  return __builtin_shufflevector(a, b, 0,1,2,3,4,5,6,7,8,9,10,11,12,13,14,15);
}

// ---------------- Kernel 1: normalize, bf16 hi/lo split, pos[] ----------------
__global__ void __launch_bounds__(256)
prep_kernel(const float* __restrict__ p1, const float* __restrict__ p2,
            bf16* __restrict__ zhi, bf16* __restrict__ zlo,
            float* __restrict__ pos) {
  const int r = blockIdx.x;
  const int tid = threadIdx.x;
  const float4 a = ((const float4*)(p1 + (size_t)r * DIM))[tid];
  const float4 b = ((const float4*)(p2 + (size_t)r * DIM))[tid];
  float s1 = a.x*a.x + a.y*a.y + a.z*a.z + a.w*a.w;
  float s2 = b.x*b.x + b.y*b.y + b.z*b.z + b.w*b.w;
  float d  = a.x*b.x + a.y*b.y + a.z*b.z + a.w*b.w;
#pragma unroll
  for (int m = 1; m < 32; m <<= 1) {
    s1 += __shfl_xor(s1, m, 32);
    s2 += __shfl_xor(s2, m, 32);
    d  += __shfl_xor(d,  m, 32);
  }
  __shared__ float sh[3][8];
  __shared__ float inv12[2];
  const int lane = tid & 31, wid = tid >> 5;
  if (lane == 0) { sh[0][wid] = s1; sh[1][wid] = s2; sh[2][wid] = d; }
  __syncthreads();
  if (tid == 0) {
    float t1 = 0.f, t2 = 0.f, td = 0.f;
    for (int i = 0; i < 8; ++i) { t1 += sh[0][i]; t2 += sh[1][i]; td += sh[2][i]; }
    const float i1 = rsqrtf(t1), i2 = rsqrtf(t2);
    inv12[0] = i1; inv12[1] = i2;
    pos[r] = td * i1 * i2;  // z_i . z_j for this row
  }
  __syncthreads();
  const float i1 = inv12[0], i2 = inv12[1];
  const size_t o1 = (size_t)r * DIM + tid * 4;
  const size_t o2 = (size_t)(r + N_ROWS) * DIM + tid * 4;
  const float va[4] = {a.x*i1, a.y*i1, a.z*i1, a.w*i1};
  const float vb[4] = {b.x*i2, b.y*i2, b.z*i2, b.w*i2};
#pragma unroll
  for (int j = 0; j < 4; ++j) {
    bf16 h = (bf16)va[j];
    zhi[o1 + j] = h;
    zlo[o1 + j] = (bf16)(va[j] - (float)h);
    bf16 g = (bf16)vb[j];
    zhi[o2 + j] = g;
    zlo[o2 + j] = (bf16)(vb[j] - (float)g);
  }
}

// ---------------- Kernel 2: fused WMMA GEMM + softmax-denominator + unif ------
__global__ void __launch_bounds__(256)
sim_gemm_kernel(const bf16* __restrict__ zhi, const bf16* __restrict__ zlo,
                float* __restrict__ pdenom, float* __restrict__ upart) {
  __shared__ __align__(16) bf16 sAhi[2][BM][KB];
  __shared__ __align__(16) bf16 sAlo[2][BM][KB];
  __shared__ __align__(16) bf16 sBhi[2][BN][KB];
  __shared__ __align__(16) bf16 sBlo[2][BN][KB];

  const int tid   = threadIdx.x;
  const int lane  = tid & 31;
  const int wid   = tid >> 5;
  const int wr    = wid >> 2;   // wave row band (0..1), 32 rows each
  const int wc    = wid & 3;    // wave col band (0..3), 32 cols each
  const int nlane = lane & 15;
  const int khalf = lane >> 4;
  const int row0  = blockIdx.y * BM;
  const int col0  = blockIdx.x * BN;

  f32x8 acc[2][2] = {};

  // Copy-thread mapping: row = tid>>2 (0..63), c4 = tid&3 (16B chunk within a
  // 64B k-slice). 6 straight-line async issues per thread, bases are kernel
  // args (SGPR-uniform), no divergent region selection.
  const int crow = tid >> 2;
  const int c4   = tid & 3;

  auto issue = [&](int buf, int k) {
    const uint32_t ga  = (uint32_t)(((row0 + crow) * DIM + k + c4 * 8) * 2);
    const uint32_t gb0 = (uint32_t)(((col0 + crow) * DIM + k + c4 * 8) * 2);
    const uint32_t gb1 = (uint32_t)(((col0 + 64 + crow) * DIM + k + c4 * 8) * 2);
    async_copy_b128(lds_addr(&sAhi[buf][crow][c4 * 8]),      zhi, ga);
    async_copy_b128(lds_addr(&sAlo[buf][crow][c4 * 8]),      zlo, ga);
    async_copy_b128(lds_addr(&sBhi[buf][crow][c4 * 8]),      zhi, gb0);
    async_copy_b128(lds_addr(&sBhi[buf][64 + crow][c4 * 8]), zhi, gb1);
    async_copy_b128(lds_addr(&sBlo[buf][crow][c4 * 8]),      zlo, gb0);
    async_copy_b128(lds_addr(&sBlo[buf][64 + crow][c4 * 8]), zlo, gb1);
  };

  issue(0, 0);  // prologue: 6 async loads in flight per wave

  for (int ks = 0; ks < NKSTEP; ++ks) {
    const int buf = ks & 1;
    if (ks + 1 < NKSTEP) {
      issue(buf ^ 1, (ks + 1) * KB);               // 12 outstanding
      asm volatile("s_wait_asynccnt 0x6" ::: "memory");  // oldest 6 done
    } else {
      asm volatile("s_wait_asynccnt 0x0" ::: "memory");
    }
    __syncthreads();   // all waves' async writes to `buf` visible

    // A 16x32 bf16 frag: lane m=nlane; v0-3: K=khalf*8+0..7, v4-7: +16.
    // B 32x16 bf16 frag: lane n=nlane; v0-3: K=khalf*8+0..7, v4-7: +16.
    bf16x16 ah[2], al[2], bh[2], bl[2];
#pragma unroll
    for (int mi = 0; mi < 2; ++mi) {
      const int am = wr * 32 + mi * 16 + nlane;
      ah[mi] = comb(*(const bf16x8*)&sAhi[buf][am][khalf * 8],
                    *(const bf16x8*)&sAhi[buf][am][16 + khalf * 8]);
      al[mi] = comb(*(const bf16x8*)&sAlo[buf][am][khalf * 8],
                    *(const bf16x8*)&sAlo[buf][am][16 + khalf * 8]);
    }
#pragma unroll
    for (int ni = 0; ni < 2; ++ni) {
      const int bn = wc * 32 + ni * 16 + nlane;
      bh[ni] = comb(*(const bf16x8*)&sBhi[buf][bn][khalf * 8],
                    *(const bf16x8*)&sBhi[buf][bn][16 + khalf * 8]);
      bl[ni] = comb(*(const bf16x8*)&sBlo[buf][bn][khalf * 8],
                    *(const bf16x8*)&sBlo[buf][bn][16 + khalf * 8]);
    }
    // s ~= hi*hi + hi*lo + lo*hi (fp32-class precision from bf16 WMMA)
#pragma unroll
    for (int mi = 0; mi < 2; ++mi)
#pragma unroll
      for (int ni = 0; ni < 2; ++ni) {
        acc[mi][ni] = __builtin_amdgcn_wmma_f32_16x16x32_bf16(
            false, ah[mi], false, bh[ni], (short)0, acc[mi][ni], false, false);
        acc[mi][ni] = __builtin_amdgcn_wmma_f32_16x16x32_bf16(
            false, ah[mi], false, bl[ni], (short)0, acc[mi][ni], false, false);
        acc[mi][ni] = __builtin_amdgcn_wmma_f32_16x16x32_bf16(
            false, al[mi], false, bh[ni], (short)0, acc[mi][ni], false, false);
      }
    __syncthreads();   // all waves done reading `buf` before it is refilled
  }

  // Fused epilogue: exp(sim/t) row sums (diag masked) + uniformity sums.
  float u0 = 0.f, u1 = 0.f;
  const int colchunk = blockIdx.x * 4 + wc;
#pragma unroll
  for (int mi = 0; mi < 2; ++mi) {
#pragma unroll
    for (int v = 0; v < 8; ++v) {
      const int grow = row0 + wr * 32 + mi * 16 + v + 8 * khalf;
      float rs = 0.f;
#pragma unroll
      for (int ni = 0; ni < 2; ++ni) {
        const int gcol = col0 + wc * 32 + ni * 16 + nlane;
        const float s = acc[mi][ni][v];
        rs += (grow == gcol) ? 0.f : __expf(10.0f * s);
        // unit rows: ||zi-zj||^2 = 2-2s, so exp(-2*d2) = exp(4s-4)
        const bool inA = (grow < 2048) && (gcol < 2048);
        const bool inB = (grow >= 2048) && (grow < 4096) &&
                         (gcol >= 2048) && (gcol < 4096);
        if ((inA || inB) && (gcol > grow)) {
          const float eu = __expf(4.0f * s - 4.0f);
          if (inA) u0 += eu; else u1 += eu;
        }
      }
      rs += __shfl_xor(rs, 1, 32);
      rs += __shfl_xor(rs, 2, 32);
      rs += __shfl_xor(rs, 4, 32);
      rs += __shfl_xor(rs, 8, 32);
      if (nlane == 0)
        pdenom[(size_t)grow * COLCHUNKS + colchunk] = rs;  // deterministic
    }
  }
#pragma unroll
  for (int m = 1; m < 32; m <<= 1) {
    u0 += __shfl_xor(u0, m, 32);
    u1 += __shfl_xor(u1, m, 32);
  }
  if (lane == 0) {
    const int wgid = blockIdx.y * gridDim.x + blockIdx.x;
    upart[(size_t)(wgid * 8 + wid) * 2 + 0] = u0;
    upart[(size_t)(wgid * 8 + wid) * 2 + 1] = u1;
  }
}

// ---------------- Kernel 3: deterministic final reduction ---------------------
__global__ void __launch_bounds__(256)
finalize_kernel(const float* __restrict__ pdenom, const float* __restrict__ pos,
                const float* __restrict__ upart, float* __restrict__ out) {
  const int tid = threadIdx.x;
  float tlog = 0.f, tpos = 0.f, talign = 0.f, tu0 = 0.f, tu1 = 0.f;
  for (int r = tid; r < ZROWS; r += 256) {
    const float* row = pdenom + (size_t)r * COLCHUNKS;
    float s = 0.f;
    for (int c = 0; c < COLCHUNKS; ++c) s += row[c];
    tlog += __logf(s);
  }
  for (int i = tid; i < N_ROWS; i += 256) {
    const float p = pos[i];
    tpos += p;
    talign += 2.f - 2.f * p;
  }
  for (int i = tid; i < ZROWS * 8; i += 256) {
    tu0 += upart[2 * i + 0];
    tu1 += upart[2 * i + 1];
  }
#pragma unroll
  for (int m = 1; m < 32; m <<= 1) {
    tlog += __shfl_xor(tlog, m, 32);
    tpos += __shfl_xor(tpos, m, 32);
    talign += __shfl_xor(talign, m, 32);
    tu0 += __shfl_xor(tu0, m, 32);
    tu1 += __shfl_xor(tu1, m, 32);
  }
  __shared__ float sh[5][8];
  const int lane = tid & 31, wid = tid >> 5;
  if (lane == 0) {
    sh[0][wid] = tlog; sh[1][wid] = tpos; sh[2][wid] = talign;
    sh[3][wid] = tu0;  sh[4][wid] = tu1;
  }
  __syncthreads();
  if (tid == 0) {
    float L = 0.f, P = 0.f, A = 0.f, U0 = 0.f, U1 = 0.f;
    for (int i = 0; i < 8; ++i) {
      L += sh[0][i]; P += sh[1][i]; A += sh[2][i]; U0 += sh[3][i]; U1 += sh[4][i];
    }
    // sum positives = 2*P ; /t = *10
    const float loss   = (L - 20.0f * P) / (float)ZROWS;
    const float lalign = A / (float)N_ROWS;
    const float npairs = 2048.0f * 2047.0f * 0.5f;
    const float lunif  = 0.5f * (__logf(U0 / npairs) + __logf(U1 / npairs));
    out[0] = loss; out[1] = lalign; out[2] = lunif;
  }
}

extern "C" void kernel_launch(void* const* d_in, const int* in_sizes, int n_in,
                              void* d_out, int out_size, void* d_ws, size_t ws_size,
                              hipStream_t stream) {
  (void)in_sizes; (void)n_in; (void)out_size; (void)ws_size;
  const float* p1 = (const float*)d_in[0];
  const float* p2 = (const float*)d_in[1];
  float* out = (float*)d_out;

  char* ws = (char*)d_ws;
  bf16*  zhi    = (bf16*)ws;                                  // 16 MB
  bf16*  zlo    = (bf16*)(ws + (size_t)ZROWS * DIM * 2);      // 16 MB
  float* pos    = (float*)(ws + (size_t)ZROWS * DIM * 4);     // 16 KB
  float* pdenom = pos + N_ROWS;                               // 8 MB
  float* upart  = pdenom + (size_t)ZROWS * COLCHUNKS;         // 512 KB

  prep_kernel<<<N_ROWS, 256, 0, stream>>>(p1, p2, zhi, zlo, pos);
  dim3 grid(ZROWS / BN, ZROWS / BM);  // (64, 128)
  sim_gemm_kernel<<<grid, 256, 0, stream>>>(zhi, zlo, pdenom, upart);
  finalize_kernel<<<1, 256, 0, stream>>>(pdenom, pos, upart, out);
}
